// EdgeDecoder_81801947120312
// MI455X (gfx1250) — matrix operand order repaired and checked by reading, not compiled, not run on hardware
//
#include <hip/hip_runtime.h>
#include <hip/hip_bf16.h>

typedef __attribute__((ext_vector_type(16))) __bf16 v16bf;
typedef __attribute__((ext_vector_type(8)))  float  v8f;

#define H 128
#define LDSPITCH 132   // 128-dword row + 4-dword pad -> conflict-free b128 fragment loads

__device__ __forceinline__ unsigned pack_bf16_rne(float a, float b) {
    unsigned ua = __builtin_bit_cast(unsigned, a);
    unsigned ub = __builtin_bit_cast(unsigned, b);
    ua += 0x7FFFu + ((ua >> 16) & 1u);   // round-to-nearest-even
    ub += 0x7FFFu + ((ub >> 16) & 1u);
    return (ua >> 16) | (ub & 0xFFFF0000u);
}

union Frag {
    unsigned u[8];
    uint4    q[2];
    v16bf    v;
};

__global__ void __launch_bounds__(256)
edge_decoder_kernel(const float* __restrict__ z_user,
                    const float* __restrict__ z_anime,
                    const int*   __restrict__ row,
                    const int*   __restrict__ col,
                    const float* __restrict__ W1,
                    const float* __restrict__ b1,
                    const float* __restrict__ W2,
                    const float* __restrict__ b2,
                    float*       __restrict__ out,
                    int n_edge, int n_tile)
{
    __shared__ __align__(16) unsigned atile[16 * LDSPITCH]; // 16 edges x 256 bf16 (padded)
    __shared__ float accs[16];

    const int tid  = threadIdx.x;   // 0..255
    const int wave = tid >> 5;      // 0..7   -> N-tile (cols wave*16..+15)
    const int lane = tid & 31;

    // ---- Preload this wave's W1 slab as bf16 B-fragments, resident in VGPRs for all tiles.
    // B fragments in this code's contiguous K order (32x16): lanes 0-15 hold K=0..15 (pairs in v0..v7),
    // lanes 16-31 hold K=16..31. lane -> column n, VGPR i -> K pair.
    const int n  = wave * 16 + (lane & 15);   // output column this lane owns
    const int kb = (lane >> 4) * 16;          // K half offset within 32-wide step
    Frag bfrag[8];
    #pragma unroll
    for (int kk = 0; kk < 8; ++kk) {
        #pragma unroll
        for (int i = 0; i < 8; ++i) {
            // lanes 0-15 read 64B-contiguous (consecutive n) -> coalesced, L2-resident
            float f0 = W1[(kk * 32 + kb + 2 * i    ) * H + n];
            float f1 = W1[(kk * 32 + kb + 2 * i + 1) * H + n];
            bfrag[kk].u[i] = pack_bf16_rne(f0, f1);
        }
    }
    const float w2v = W2[n];
    const float b1v = b1[n];
    const float b2v = b2[0];

    // A-fragment addressing (16-bit A 16x32): lane -> row r = lane&15,
    // lanes 0-15: K 0..15 in v0..7; lanes 16-31: K 16..31 (contiguous, same as B).
    const int r  = lane & 15;
    const int kh = lane >> 4;
    const unsigned* arow = atile + r * LDSPITCH + kh * 8;

    for (int tile = blockIdx.x; tile < n_tile; tile += gridDim.x) {
        const int e0 = tile * 16;
        __syncthreads();                     // previous tile's readers are done
        if (tid < 16) accs[tid] = 0.0f;

        // ---- Gather + convert: 16 edges x (128 zu | 128 za) -> bf16 A tile in LDS.
        {
            const int el = tid >> 4;         // local edge 0..15
            const int ch = tid & 15;         // 16-float chunk 0..15
            int e  = e0 + el;
            int ec = e < n_edge ? e : n_edge - 1;
            const float* src = (ch < 8)
                ? z_user  + (size_t)row[ec] * H + ch * 16
                : z_anime + (size_t)col[ec] * H + (ch - 8) * 16;
            const float4* s4 = (const float4*)src;   // 64B-aligned
            float4 f0 = s4[0], f1 = s4[1], f2 = s4[2], f3 = s4[3];
            unsigned* d = atile + el * LDSPITCH + ch * 8;
            *(uint4*)(d)     = make_uint4(pack_bf16_rne(f0.x, f0.y), pack_bf16_rne(f0.z, f0.w),
                                          pack_bf16_rne(f1.x, f1.y), pack_bf16_rne(f1.z, f1.w));
            *(uint4*)(d + 4) = make_uint4(pack_bf16_rne(f2.x, f2.y), pack_bf16_rne(f2.z, f2.w),
                                          pack_bf16_rne(f3.x, f3.y), pack_bf16_rne(f3.z, f3.w));

            // Prefetch next tile's random gather rows into near caches (WGP scope).
            int tn = tile + gridDim.x;
            if (tn < n_tile) {
                int en = tn * 16 + el;
                if (en >= n_edge) en = n_edge - 1;
                const float* pn = (ch < 8)
                    ? z_user  + (size_t)row[en] * H + ch * 16
                    : z_anime + (size_t)col[en] * H + (ch - 8) * 16;
                __builtin_prefetch(pn, 0, 3);
            }
        }
        __syncthreads();

        // ---- 16x16 h-tile per wave: D = A(16x256) x W1slab(256x16), K in 8 steps of 32.
        // Distinct fragment registers per step so all 16 ds_load_b128 issue up front
        // and the 8 WMMAs chain without full dscnt drains between them.
        Frag a[8];
        #pragma unroll
        for (int kk = 0; kk < 8; ++kk) {
            a[kk].q[0] = *(const uint4*)(arow + kk * 16);      // K 0..7  (this half, +16h)
            a[kk].q[1] = *(const uint4*)(arow + kk * 16 + 4);  // K 8..15 (this half, +16h)
        }
        v8f c = {};
        #pragma unroll
        for (int kk = 0; kk < 8; ++kk) {
            c = __builtin_amdgcn_wmma_f32_16x16x32_bf16(
                    false, a[kk].v, false, bfrag[kk].v, (short)0, c, false, false);
        }

        // ---- Epilogue: +b1, ReLU, *W2[col]; reduce 16 cols -> per-edge scalar.
        // C/D layout: lane&15 = N, VGPR j -> edge M = j + 8*(lane>>4).
        #pragma unroll
        for (int j = 0; j < 8; ++j) {
            float h = c[j] + b1v;
            h = h > 0.0f ? h : 0.0f;
            float v = h * w2v;
            v += __shfl_xor(v, 1, 32);
            v += __shfl_xor(v, 2, 32);
            v += __shfl_xor(v, 4, 32);
            v += __shfl_xor(v, 8, 32);
            if ((lane & 15) == 0)
                atomicAdd(&accs[(lane >> 4) * 8 + j], v);  // ds_add_f32 across waves
        }
        __syncthreads();

        if (tid < 16) {
            int e = e0 + tid;
            if (e < n_edge) out[e] = accs[tid] + b2v;
        }
    }
}

extern "C" void kernel_launch(void* const* d_in, const int* in_sizes, int n_in,
                              void* d_out, int out_size, void* d_ws, size_t ws_size,
                              hipStream_t stream) {
    const float* z_user  = (const float*)d_in[0];
    const float* z_anime = (const float*)d_in[1];
    const int*   row     = (const int*)d_in[2];
    const int*   col     = (const int*)d_in[3];
    const float* W1      = (const float*)d_in[4];
    const float* b1      = (const float*)d_in[5];
    const float* W2      = (const float*)d_in[6];
    const float* b2      = (const float*)d_in[7];
    float* out = (float*)d_out;

    const int n_edge = in_sizes[2];
    const int n_tile = (n_edge + 15) / 16;
    int blocks = n_tile < 2048 ? n_tile : 2048;
    if (blocks < 1) blocks = 1;

    edge_decoder_kernel<<<blocks, 256, 0, stream>>>(
        z_user, z_anime, row, col, W1, b1, W2, b2, out, n_edge, n_tile);
}
